// TemporalSelfAttention_42288247996572
// MI455X (gfx1250) — matrix operand order repaired and checked by reading, not compiled, not run on hardware
//
#include <hip/hip_runtime.h>
#include <hip/hip_bf16.h>
#include <math.h>

// ---------------------------------------------------------------------------
// Linear (kernelized) attention, MI455X / gfx1250, wave32 + WMMA f16->f32.
//   f=1024 frames, n=64 joints, c=256, hid=1024, heads=8, d=128
// Stage 0: convert x -> f16, W -> f16 transposed (hid x c)
// Stage 1: Q/K/V = xt@W + b, elu+1 on Q,K  (WMMA; B tile async-staged in LDS)
// Stage 2: KV[m][d] = sum_s V[s][m]*K[s][d], Ksum[d] (WMMA via LDS transpose)
// Stage 3: out[l][m] = Z[l]*sum_d Q[l][d]*KV[m][d] (WMMA; KV async-staged LDS)
// ---------------------------------------------------------------------------

#define FDIM 1024
#define NDIM 64
#define CDIM 256
#define HID  1024
#define HEADS 8
#define DDIM 128
#define NH   (NDIM * HEADS)

typedef __attribute__((ext_vector_type(16))) _Float16 v16h;
typedef __attribute__((ext_vector_type(8)))  _Float16 v8h;
typedef __attribute__((ext_vector_type(8)))  float    v8f;

union AV { v16h v; v8h h[2]; };

__device__ __forceinline__ v8f wmma_f16(v16h a, v16h b, v8f c) {
  // (neg_a, A, neg_b, B, c_mod, C, reuse_a, reuse_b)
  return __builtin_amdgcn_wmma_f32_16x16x32_f16(false, a, false, b, (short)0, c,
                                                false, false);
}

// gfx1250 async global->LDS copy, 16B per active lane. Tracked by ASYNCcnt.
// lds_off: byte offset within the workgroup's LDS segment (generic shared
// pointers map to LDS by truncation to the low 32 bits, ISA 10.2).
__device__ __forceinline__ void async_copy_b128(unsigned lds_off,
                                                const void* gaddr) {
  asm volatile("global_load_async_to_lds_b128 %0, %1, off"
               :: "v"(lds_off), "v"((unsigned long long)(uintptr_t)gaddr)
               : "memory");
}
__device__ __forceinline__ void wait_async0() {
  asm volatile("s_wait_asynccnt 0" ::: "memory");
}

// ---------------- Stage 0: conversions -------------------------------------

__global__ __launch_bounds__(256) void cvt_x_kernel(const float* __restrict__ x,
                                                    _Float16* __restrict__ xh,
                                                    size_t nelem) {
  for (size_t i = (size_t)blockIdx.x * blockDim.x + threadIdx.x; i < nelem;
       i += (size_t)gridDim.x * blockDim.x)
    xh[i] = (_Float16)x[i];
}

// Wt[z][col][k] = W_z[k][col]   (z in {q,k,v}), 3*HID*CDIM halves
__global__ __launch_bounds__(256) void cvt_w_kernel(const float* __restrict__ Wq,
                                                    const float* __restrict__ Wk,
                                                    const float* __restrict__ Wv,
                                                    _Float16* __restrict__ Wt) {
  int idx = blockIdx.x * blockDim.x + threadIdx.x;   // 3*1024*256 total
  int k    = idx & (CDIM - 1);
  int col  = (idx >> 8) & (HID - 1);
  int z    = idx >> 18;
  const float* W = (z == 0) ? Wq : (z == 1) ? Wk : Wv;
  Wt[idx] = (_Float16)W[(size_t)k * HID + col];
}

// ---------------- Stage 1: fused QKV projection + elu feature map ----------
// A rows r = n*F + f map to x[f][n][:] ; output row-major (N*F, HID) f16.
// B tile (64 cols x 256 K = 32KB) is async-staged into LDS once per block.

#define BSTRIDE 264   // halves per LDS B row: 528B, 16B-aligned, bank-skewed

__global__ __launch_bounds__(256) void qkv_proj_kernel(
    const _Float16* __restrict__ xh,   // (F, N, C)
    const _Float16* __restrict__ Wt,   // (3, HID, C) transposed weights
    const float* __restrict__ bq, const float* __restrict__ bk,
    const float* __restrict__ bv,
    _Float16* __restrict__ Qf, _Float16* __restrict__ Kf,
    _Float16* __restrict__ Vf) {
  __shared__ _Float16 Bl[64 * BSTRIDE];           // 16,896 halves

  const int wave = threadIdx.x >> 5;
  const int lane = threadIdx.x & 31;
  const int z = blockIdx.z;                       // 0=q 1=k 2=v
  const int mbase = blockIdx.x * 128 + wave * 16; // output row tile
  const int nbase = blockIdx.y * 64;              // output col tile (4x16)

  const _Float16* W = Wt + (size_t)z * HID * CDIM;
  const float* bias = (z == 0) ? bq : (z == 1) ? bk : bv;
  _Float16* Out = (z == 0) ? Qf : (z == 1) ? Kf : Vf;

  // async-stage B tile: source is 64 contiguous rows of 512B at W+nbase*CDIM
  {
    const _Float16* src = W + (size_t)nbase * CDIM;
    const unsigned lds0 = (unsigned)(uintptr_t)(&Bl[0]);
#pragma unroll
    for (int i = 0; i < 8; ++i) {
      const int c = threadIdx.x + 256 * i;        // 16B chunk id, 2048 total
      const int row = c >> 5;                      // col index 0..63
      const int off = (c & 31) * 8;                // halves within row
      async_copy_b128(lds0 + (unsigned)(row * BSTRIDE + off) * 2,
                      (const void*)(src + (size_t)c * 8));
    }
  }

  // per-lane A row: r -> (n = r>>10, f = r&1023) -> xh[(f*N + n)*C]
  const int r = mbase + (lane & 15);
  const int ff = r & (FDIM - 1);
  const int nn = r >> 10;
  const int asel = lane >> 4;
  const _Float16* arow = xh + ((size_t)ff * NDIM + nn) * CDIM + asel * 8;

  wait_async0();
  __syncthreads();

  v8f acc[4] = {v8f{}, v8f{}, v8f{}, v8f{}};

#pragma unroll
  for (int kt = 0; kt < CDIM / 32; ++kt) {
    const int k0 = kt * 32;
    if (kt + 1 < CDIM / 32)
      __builtin_prefetch((const void*)(arow + k0 + 32), 0, 3);
    AV av;
    av.h[0] = *(const v8h*)(arow + k0);        // K = 8*asel .. +7
    av.h[1] = *(const v8h*)(arow + k0 + 16);   // K = 16 + 8*asel .. +7
#pragma unroll
    for (int nt = 0; nt < 4; ++nt) {
      const _Float16* bp =
          Bl + (nt * 16 + (lane & 15)) * BSTRIDE + k0 + asel * 8;
      AV bv_; bv_.h[0] = *(const v8h*)bp; bv_.h[1] = *(const v8h*)(bp + 16);
      v16h b = bv_.v;                          // lane=col, K = 8*asel + i, 16 + 8*asel + i
      acc[nt] = wmma_f16(av.v, b, acc[nt]);
    }
  }

  // epilogue: bias + elu(x)+1 for q,k ; f16 store, row-major (N*F, HID)
#pragma unroll
  for (int nt = 0; nt < 4; ++nt) {
    const int col = nbase + nt * 16 + (lane & 15);
    const float bb = bias[col];
#pragma unroll
    for (int j = 0; j < 8; ++j) {
      const int row = mbase + asel * 8 + j;    // C/D layout: VGPR j, lane-half
      float p = acc[nt][j] + bb;
      if (z != 2) p = (p > 0.f) ? (p + 1.f) : __expf(p);   // elu(p)+1
      Out[(size_t)row * HID + col] = (_Float16)p;
    }
  }
}

// ---------------- Stage 2: KV = V^T @ K  (per n,h) + Ksum ------------------

#define TSTRIDE 40   // halves; rows 80B apart -> 16B aligned, conflict-skewed

__global__ __launch_bounds__(256) void kv_kernel(
    const _Float16* __restrict__ Kf, const _Float16* __restrict__ Vf,
    _Float16* __restrict__ KVt,   // (NH, 128 m, 128 d) row-major f16
    float* __restrict__ Ksum) {   // (NH, 128)
  __shared__ _Float16 Ktl[DDIM * TSTRIDE];   // K^T tile: [d][s']
  __shared__ _Float16 Vtl[DDIM * TSTRIDE];   // V^T tile: [m][s']

  const int nh = blockIdx.x;
  const int nn = nh >> 3, h = nh & 7;
  const int wave = threadIdx.x >> 5;
  const int lane = threadIdx.x & 31;
  const int asel = lane >> 4;

  const _Float16* Kbase = Kf + (size_t)nn * FDIM * HID + h * DDIM;
  const _Float16* Vbase = Vf + (size_t)nn * FDIM * HID + h * DDIM;

  v8f acc[8] = {v8f{}, v8f{}, v8f{}, v8f{}, v8f{}, v8f{}, v8f{}, v8f{}};

  for (int s0 = 0; s0 < FDIM; s0 += 32) {
    __syncthreads();
    // stage 32x128 chunks of K and V into LDS, transposed.
    // thread t: row s' = t>>3, 16-wide d segment = (t&7)*16
    {
      const int sp = threadIdx.x >> 3;
      const int d0 = (threadIdx.x & 7) * 16;
      v16h kv_ = *(const v16h*)(Kbase + (size_t)(s0 + sp) * HID + d0);
      v16h vv_ = *(const v16h*)(Vbase + (size_t)(s0 + sp) * HID + d0);
#pragma unroll
      for (int i = 0; i < 16; ++i) {
        Ktl[(d0 + i) * TSTRIDE + sp] = kv_[i];
        Vtl[(d0 + i) * TSTRIDE + sp] = vv_[i];
      }
    }
    __syncthreads();

    // A = V^T rows m = wave*16.., K dim = s' (32)
    const int am = wave * 16 + (lane & 15);
    AV av;
    av.h[0] = *(const v8h*)(Vtl + am * TSTRIDE + asel * 8);
    av.h[1] = *(const v8h*)(Vtl + am * TSTRIDE + 16 + asel * 8);
#pragma unroll
    for (int nt = 0; nt < 8; ++nt) {
      const _Float16* bp = Ktl + (nt * 16 + (lane & 15)) * TSTRIDE + asel * 8;
      AV bv_; bv_.h[0] = *(const v8h*)bp; bv_.h[1] = *(const v8h*)(bp + 16);
      v16h b = bv_.v;
      acc[nt] = wmma_f16(av.v, b, acc[nt]);
    }
  }

  // KV[m][d], row-major f16 (exactly the B^T layout stage 3 wants)
  _Float16* kvout = KVt + (size_t)nh * DDIM * DDIM;
#pragma unroll
  for (int nt = 0; nt < 8; ++nt) {
    const int d = nt * 16 + (lane & 15);
#pragma unroll
    for (int j = 0; j < 8; ++j) {
      const int m = wave * 16 + asel * 8 + j;
      kvout[m * DDIM + d] = (_Float16)acc[nt][j];
    }
  }

  // Ksum[d] = sum_s K[s][d]
  if (threadIdx.x < DDIM) {
    const int d = threadIdx.x;
    float s = 0.f;
    for (int sp = 0; sp < FDIM; ++sp) s += (float)Kbase[(size_t)sp * HID + d];
    Ksum[(size_t)nh * DDIM + d] = s;
  }
}

// ---------------- Stage 3: out = Z * (Q @ KV^T), scatter to (f, n, hid) ----

#define OSTRIDE 136  // halves; 272B rows, 16B-aligned, bank-skewed

__global__ __launch_bounds__(256) void out_kernel(
    const _Float16* __restrict__ Qf, const _Float16* __restrict__ KVt,
    const float* __restrict__ Ksum, float* __restrict__ out) {
  __shared__ _Float16 Btile[DDIM * OSTRIDE];   // KV tile [m][d], padded
  __shared__ float zsh[128];

  const int nh = blockIdx.x;
  const int nn = nh >> 3, h = nh & 7;
  const int l0 = blockIdx.y * 128;
  const int wave = threadIdx.x >> 5;
  const int lane = threadIdx.x & 31;
  const int asel = lane >> 4;

  const _Float16* Qbase = Qf + ((size_t)nn * FDIM + l0) * HID + h * DDIM;
  const _Float16* Bt = KVt + (size_t)nh * DDIM * DDIM;   // Bt[m][d], 32KB

  // async-stage KV tile into LDS (dense 32KB source, padded rows in LDS)
  {
    const unsigned lds0 = (unsigned)(uintptr_t)(&Btile[0]);
#pragma unroll
    for (int i = 0; i < 8; ++i) {
      const int c = threadIdx.x + 256 * i;   // 16B chunk id, 2048 total
      const int row = c >> 4;                // m 0..127
      const int off = (c & 15) * 8;          // halves within row
      async_copy_b128(lds0 + (unsigned)(row * OSTRIDE + off) * 2,
                      (const void*)(Bt + (size_t)c * 8));
    }
  }

  // Z = 1/(Q . Ksum + eps) per row (overlaps with async copy)
  if (threadIdx.x < 128) {
    const _Float16* qrow = Qbase + (size_t)threadIdx.x * HID;
    const float* ks = Ksum + (size_t)nh * DDIM;
    float s = 0.f;
#pragma unroll 8
    for (int d = 0; d < DDIM; ++d) s += (float)qrow[d] * ks[d];
    zsh[threadIdx.x] = 1.f / (s + 1e-6f);
  }

  wait_async0();
  __syncthreads();

  v8f acc[8] = {v8f{}, v8f{}, v8f{}, v8f{}, v8f{}, v8f{}, v8f{}, v8f{}};
  const _Float16* arow = Qbase + (size_t)(wave * 16 + (lane & 15)) * HID + asel * 8;

#pragma unroll
  for (int kt = 0; kt < DDIM / 32; ++kt) {
    const int k0 = kt * 32;
    AV av;
    av.h[0] = *(const v8h*)(arow + k0);
    av.h[1] = *(const v8h*)(arow + k0 + 16);
#pragma unroll
    for (int nt = 0; nt < 8; ++nt) {
      const _Float16* bp =
          Btile + (nt * 16 + (lane & 15)) * OSTRIDE + k0 + asel * 8;
      AV bv_; bv_.h[0] = *(const v8h*)bp; bv_.h[1] = *(const v8h*)(bp + 16);
      v16h b = bv_.v;
      acc[nt] = wmma_f16(av.v, b, acc[nt]);
    }
  }

  // out shape (f, n, hid): out[l][nn][h*128 + m]
#pragma unroll
  for (int nt = 0; nt < 8; ++nt) {
    const int m = nt * 16 + (lane & 15);
#pragma unroll
    for (int j = 0; j < 8; ++j) {
      const int lr = wave * 16 + asel * 8 + j;
      out[((size_t)(l0 + lr) * NDIM + nn) * HID + h * DDIM + m] =
          acc[nt][j] * zsh[lr];
    }
  }
}

// ---------------------------------------------------------------------------

extern "C" void kernel_launch(void* const* d_in, const int* in_sizes, int n_in,
                              void* d_out, int out_size, void* d_ws,
                              size_t ws_size, hipStream_t stream) {
  (void)in_sizes; (void)n_in; (void)out_size; (void)ws_size;
  const float* x  = (const float*)d_in[0];
  const float* Wq = (const float*)d_in[1];
  const float* bq = (const float*)d_in[2];
  const float* Wk = (const float*)d_in[3];
  const float* bk = (const float*)d_in[4];
  const float* Wv = (const float*)d_in[5];
  const float* bv = (const float*)d_in[6];
  float* out = (float*)d_out;

  char* ws = (char*)d_ws;
  const size_t xh_elems = (size_t)FDIM * NDIM * CDIM;       // 16.7M
  const size_t wt_elems = (size_t)3 * HID * CDIM;           // 786k
  const size_t p_elems  = (size_t)NDIM * FDIM * HID;        // 67.1M each
  const size_t kv_elems = (size_t)NH * DDIM * DDIM;         // 8.4M

  _Float16* xh = (_Float16*)ws;  ws += xh_elems * sizeof(_Float16);
  _Float16* Wt = (_Float16*)ws;  ws += wt_elems * sizeof(_Float16);
  _Float16* Qf = (_Float16*)ws;  ws += p_elems * sizeof(_Float16);
  _Float16* Kf = (_Float16*)ws;  ws += p_elems * sizeof(_Float16);
  _Float16* Vf = (_Float16*)ws;  ws += p_elems * sizeof(_Float16);
  _Float16* KVt = (_Float16*)ws; ws += kv_elems * sizeof(_Float16);
  float* Ksum = (float*)ws;      ws += (size_t)NH * DDIM * sizeof(float);

  cvt_x_kernel<<<4096, 256, 0, stream>>>(x, xh, xh_elems);
  cvt_w_kernel<<<(int)(wt_elems / 256), 256, 0, stream>>>(Wq, Wk, Wv, Wt);

  dim3 g1(FDIM * NDIM / 128, HID / 64, 3);
  qkv_proj_kernel<<<g1, 256, 0, stream>>>(xh, Wt, bq, bk, bv, Qf, Kf, Vf);

  kv_kernel<<<NH, 256, 0, stream>>>(Kf, Vf, KVt, Ksum);

  dim3 g3(NH, FDIM / 128);
  out_kernel<<<g3, 256, 0, stream>>>(Qf, KVt, Ksum, out);
}
